// RNN_14035953124027
// MI455X (gfx1250) — matrix operand order repaired and checked by reading, not compiled, not run on hardware
//
#include <hip/hip_runtime.h>

// CDNA5 / gfx1250 fused tanh-RNN:
//   per-wave: 16 batch rows, full W_hh resident in VGPRs as bf16 B-fragments,
//   28 timesteps of v_wmma_f32_16x16x32_bf16, h kept in LDS (bf16) only for
//   the C-layout -> A-layout transpose. No cross-wave synchronization.

typedef __attribute__((ext_vector_type(16))) __bf16 v16bf;
typedef __attribute__((ext_vector_type(2)))  __bf16 bf16x2;
typedef __attribute__((ext_vector_type(8)))  float  v8f;

__device__ __forceinline__ float fast_tanh(float x) {
#if __has_builtin(__builtin_amdgcn_tanhf)
    return __builtin_amdgcn_tanhf(x);   // v_tanh_f32 (CDNA5 TRANS op)
#else
    return tanhf(x);
#endif
}

__global__ __launch_bounds__(128, 1)
void rnn_fused_wmma(const float* __restrict__ input,   // [B,28,28]
                    const float* __restrict__ W_ih,    // [128,28]
                    const float* __restrict__ W_hh,    // [128,128]
                    const float* __restrict__ b_ih,    // [128]
                    const float* __restrict__ b_hh,    // [128]
                    const float* __restrict__ W_out,   // [10,128]
                    const float* __restrict__ b_out,   // [10]
                    float* __restrict__ out)           // [B,10]
{
    constexpr int H  = 128;
    constexpr int T  = 28;
    constexpr int NI = 28;
    constexpr int NC = 10;
    constexpr int HP = 130;   // padded LDS row stride (bf16 elems): kills bank conflicts

    __shared__ __bf16 hbuf_all[4][16 * HP];   // 4 waves x (16 rows x 128 h, padded)

    const int lane = threadIdx.x & 31;
    const int wave = threadIdx.x >> 5;
    const int lrow = lane & 15;   // M (A/C rows) or N (B cols) within a 16-tile
    const int hi   = lane >> 4;   // which half-wave

    __bf16* hb = hbuf_all[wave];

    const int b0  = (blockIdx.x * 4 + wave) * 16;  // this wave's 16 batch rows
    const int row = b0 + lrow;

    // ------------------------------------------------------------------
    // W_hh -> bf16 B-fragments, resident in VGPRs.
    // B (32x16, K x N): lane holds N = lrow; VGPR v holds K = 2v + 16*hi.
    // B[k][n] = W_hh[n][k]  (h @ W_hh^T)
    // ------------------------------------------------------------------
    v16bf whh[8][4];
#pragma unroll
    for (int nt = 0; nt < 8; ++nt) {
        const int col = nt * 16 + lrow;
#pragma unroll
        for (int kt = 0; kt < 4; ++kt) {
#pragma unroll
            for (int p = 0; p < 8; ++p) {
                const int k = kt * 32 + 2 * p + hi * 16;
                float2 w = *reinterpret_cast<const float2*>(&W_hh[col * H + k]);
                whh[nt][kt][2 * p]     = (__bf16)w.x;
                whh[nt][kt][2 * p + 1] = (__bf16)w.y;
            }
        }
    }

    // W_ih -> bf16 B-fragments (K padded 28 -> 32 with zeros).
    v16bf wih[8];
#pragma unroll
    for (int nt = 0; nt < 8; ++nt) {
        const int col = nt * 16 + lrow;
#pragma unroll
        for (int p = 0; p < 8; ++p) {
            const int k = 2 * p + hi * 16;        // k in {0..30}, pairs (k,k+1)
            float2 w = make_float2(0.f, 0.f);
            if (!(hi == 1 && p >= 6))             // k >= 28 -> zero pad
                w = *reinterpret_cast<const float2*>(&W_ih[col * NI + k]);
            wih[nt][2 * p]     = (__bf16)w.x;
            wih[nt][2 * p + 1] = (__bf16)w.y;
        }
    }

    // Per-column bias (depends only on lane within C/D layout).
    float bias[8];
#pragma unroll
    for (int nt = 0; nt < 8; ++nt) {
        const int col = nt * 16 + lrow;
        bias[nt] = b_ih[col] + b_hh[col];
    }

    // h0 = 0
    for (int i = lane; i < 16 * HP; i += 32) hb[i] = (__bf16)0.f;
    asm volatile("s_wait_dscnt 0x0" ::: "memory");

    // ------------------------------------------------------------------
    // Recurrence: h = tanh( x_t @ W_ih^T + h @ W_hh^T + b )
    // ------------------------------------------------------------------
#pragma unroll 1
    for (int t = 0; t < T; ++t) {
        // input A-fragment (16x32 bf16, M=lrow):
        // VGPR v: K = 2v + 16*hi; K>=28 zero-padded.
        v16bf ain;
        const float* inrow = input + ((size_t)row * T + t) * NI;
#pragma unroll
        for (int p = 0; p < 8; ++p) {
            const int k = 2 * p + hi * 16;
            float2 v = make_float2(0.f, 0.f);
            if (!(hi == 1 && p >= 6))             // k = 28,30 -> zero (also avoids OOB)
                v = *reinterpret_cast<const float2*>(&inrow[k]);
            ain[2 * p]     = (__bf16)v.x;
            ain[2 * p + 1] = (__bf16)v.y;
        }

        // h A-fragments from LDS (4 K-tiles of 32)
        v16bf ha[4];
#pragma unroll
        for (int kt = 0; kt < 4; ++kt) {
#pragma unroll
            for (int p = 0; p < 8; ++p) {
                const int k = kt * 32 + 2 * p + hi * 16;
                bf16x2 pr = *reinterpret_cast<const bf16x2*>(&hb[lrow * HP + k]);
                ha[kt][2 * p]     = pr.x;
                ha[kt][2 * p + 1] = pr.y;
            }
        }
        asm volatile("" ::: "memory");  // all LDS reads issued before writes below

        // 8 independent 5-deep WMMA chains (one per 16-wide slice of H)
#pragma unroll
        for (int nt = 0; nt < 8; ++nt) {
            v8f c = { bias[nt], bias[nt], bias[nt], bias[nt],
                      bias[nt], bias[nt], bias[nt], bias[nt] };
            c = __builtin_amdgcn_wmma_f32_16x16x32_bf16(
                    false, ain, false, wih[nt], (short)0, c, false, false);
#pragma unroll
            for (int kt = 0; kt < 4; ++kt)
                c = __builtin_amdgcn_wmma_f32_16x16x32_bf16(
                        false, ha[kt], false, whh[nt][kt], (short)0, c, false, false);

            // tanh + write back to LDS in row-major bf16
            // C/D layout: VGPR j -> row (j + 8*hi), col = nt*16 + lrow
#pragma unroll
            for (int j = 0; j < 8; ++j) {
                const float hv = fast_tanh(c[j]);
                hb[(j + hi * 8) * HP + nt * 16 + lrow] = (__bf16)hv;
            }
        }
        asm volatile("s_wait_dscnt 0x0" ::: "memory");  // h visible for next step
    }

    // ------------------------------------------------------------------
    // Head: out = h_last @ W_out^T + b_out  (tiny, VALU)
    // lanes 0-15 -> classes 0..4, lanes 16-31 -> classes 5..9, row = lrow
    // ------------------------------------------------------------------
#pragma unroll
    for (int q = 0; q < 5; ++q) {
        const int cls = hi * 5 + q;
        float acc = b_out[cls];
        const float* wrow = W_out + cls * H;
#pragma unroll 8
        for (int k = 0; k < H; ++k)
            acc += (float)hb[lrow * HP + k] * wrow[k];
        out[(size_t)(b0 + lrow) * NC + cls] = acc;
    }
}

extern "C" void kernel_launch(void* const* d_in, const int* in_sizes, int n_in,
                              void* d_out, int out_size, void* d_ws, size_t ws_size,
                              hipStream_t stream) {
    (void)n_in; (void)d_ws; (void)ws_size; (void)out_size;
    const float* input = (const float*)d_in[0];
    const float* W_ih  = (const float*)d_in[1];
    const float* W_hh  = (const float*)d_in[2];
    const float* b_ih  = (const float*)d_in[3];
    const float* b_hh  = (const float*)d_in[4];
    const float* W_out = (const float*)d_in[5];
    const float* b_out = (const float*)d_in[6];
    float* out = (float*)d_out;

    const int B = in_sizes[0] / (28 * 28);     // 32768
    dim3 grid(B / 64), block(128);             // 4 waves/block, 16 rows/wave
    hipLaunchKernelGGL(rnn_fused_wmma, grid, block, 0, stream,
                       input, W_ih, W_hh, b_ih, b_hh, W_out, b_out, out);
}
